// VectorConditionedLogitsGMPriorNLL_61125974557574
// MI455X (gfx1250) — matrix operand order repaired and checked by reading, not compiled, not run on hardware
//
#include <hip/hip_runtime.h>
#include <hip/hip_bf16.h>

typedef __attribute__((ext_vector_type(16))) _Float16 v16h;
typedef __attribute__((ext_vector_type(16), aligned(16))) _Float16 v16h_al; // 16B-aligned LDS view
typedef __attribute__((ext_vector_type(8)))  _Float16 v8h;
typedef __attribute__((ext_vector_type(8)))  float    v8f;
typedef __attribute__((ext_vector_type(4)))  float    v4f;

#define B_  256
#define S_  1024
#define D_  32
#define K_  512

static constexpr float LOG2E    = 1.4426950408889634f;
static constexpr float LN2      = 0.6931471805599453f;
static constexpr float LOG_NORM = 29.40603306254952f;   // D * 0.5 * ln(2*pi), STD=1

__launch_bounds__(256)
__global__ void gmm_nll_wmma_kernel(const float* __restrict__ x,
                                    const float* __restrict__ logits,
                                    const float* __restrict__ means,
                                    float* __restrict__ out) {
  // LDS: means as f16, row stride 40 halves (80 B) to spread banks;
  // t2[k] = (logits[b,k] - 0.5*||mu_k||^2) * log2(e); reduce scratch; out staging.
  __shared__ _Float16 sm_means[K_ * 40];
  __shared__ float    sm_t2[K_];
  __shared__ float    sm_red[16];
  __shared__ float    sm_stage[8 * 16];

  const int tid  = threadIdx.x;
  const int lane = tid & 31;
  const int wid  = tid >> 5;
  const int b    = blockIdx.y;
  const int sblk = blockIdx.x;     // 8 blocks of 128 rows each

  // ---------- cooperative staging: means f32->f16, m2, t2, logits ----------
  float lv[2];
  #pragma unroll
  for (int rr = 0; rr < 2; ++rr) {
    const int r = tid + rr * 256;                // component row 0..511
    const float* mrow = means + r * D_;
    float m2 = 0.f;
    #pragma unroll
    for (int c = 0; c < 4; ++c) {
      v4f f0 = *(const v4f*)(mrow + c * 8);
      v4f f1 = *(const v4f*)(mrow + c * 8 + 4);
      v8h h;
      #pragma unroll
      for (int j = 0; j < 4; ++j) {
        m2 += f0[j] * f0[j] + f1[j] * f1[j];
        h[j]     = (_Float16)f0[j];
        h[j + 4] = (_Float16)f1[j];
      }
      *(v8h*)&sm_means[r * 40 + c * 8] = h;      // 16B aligned (80B row stride)
    }
    lv[rr] = logits[(size_t)b * K_ + r];
    sm_t2[r] = (lv[rr] - 0.5f * m2) * LOG2E;
  }

  // ---------- block logsumexp over logits[b,:] (base 2) ----------
  float mx = __builtin_fmaxf(lv[0], lv[1]);
  #pragma unroll
  for (int off = 16; off >= 1; off >>= 1)
    mx = __builtin_fmaxf(mx, __shfl_xor(mx, off, 32));
  if (lane == 0) sm_red[wid] = mx;
  __syncthreads();
  if (tid == 0) {
    float g = sm_red[0];
    #pragma unroll
    for (int i = 1; i < 8; ++i) g = __builtin_fmaxf(g, sm_red[i]);
    sm_red[8] = g;
  }
  __syncthreads();
  const float gmax = sm_red[8];
  float ss = __builtin_amdgcn_exp2f((lv[0] - gmax) * LOG2E) +
             __builtin_amdgcn_exp2f((lv[1] - gmax) * LOG2E);
  #pragma unroll
  for (int off = 16; off >= 1; off >>= 1)
    ss += __shfl_xor(ss, off, 32);
  if (lane == 0) sm_red[wid] = ss;
  __syncthreads();
  if (tid == 0) {
    float t = 0.f;
    #pragma unroll
    for (int i = 0; i < 8; ++i) t += sm_red[i];
    sm_red[9] = gmax * LOG2E + __builtin_amdgcn_logf(t);   // lse2_b (base 2)
  }
  __syncthreads();
  const float lse2_b = sm_red[9];

  // ---------- per-wave 16-row s-tile ----------
  const int s_base = sblk * 128 + wid * 16;
  const int row    = lane & 15;
  const int dbase  = (lane >> 4) * 16;  // A layout: lower lanes d{0..15}, upper d{16..31}
  const float* xrow = x + ((size_t)b * S_ + (s_base + row)) * D_ + dbase;

  v4f c0 = *(const v4f*)(xrow + 0);
  v4f c1 = *(const v4f*)(xrow + 4);
  v4f c2 = *(const v4f*)(xrow + 8);
  v4f c3 = *(const v4f*)(xrow + 12);

  float x2p = 0.f;
  #pragma unroll
  for (int j = 0; j < 4; ++j)
    x2p += c0[j]*c0[j] + c1[j]*c1[j] + c2[j]*c2[j] + c3[j]*c3[j];
  const float x2 = x2p + __shfl_xor(x2p, 16, 32);   // combine two half-rows

  v16h a;
  #pragma unroll
  for (int j = 0; j < 4; ++j) {
    a[j]      = (_Float16)c0[j];
    a[j + 4]  = (_Float16)c1[j];
    a[j + 8]  = (_Float16)c2[j];
    a[j + 12] = (_Float16)c3[j];
  }

  float mrun = -1e30f;
  float ssum[8];
  #pragma unroll
  for (int i = 0; i < 8; ++i) ssum[i] = 0.f;

  // B layout: each lane's operand is 32 contiguous bytes in LDS
  // (lower lanes d 0..15, upper lanes d 16..31 of component row r).
  const int bco_h = (lane < 16) ? 0 : 16;   // offset in halves
  const v8f cz = {};

  // depth-2 software pipeline: double-buffered B tile + t2
  v16h  b0 = *(const v16h_al*)&sm_means[(row +  0) * 40 + bco_h];
  v16h  b1 = *(const v16h_al*)&sm_means[(row + 16) * 40 + bco_h];
  float t0 = sm_t2[row];
  float t1 = sm_t2[row + 16];

  // One pipeline step: WMMA on current buffer, refill it for kt+2 in the
  // WMMA shadow, then logsumexp update. Prefetch is UNCONDITIONAL (index
  // wraps mod K) so the loop body stays branch-free and the compiler can
  // use precise partial s_wait_dscnt instead of conservative dscnt==0.
  auto step = [&](v16h& bslot, float& tslot, int rn) {
    const v16h  bv  = bslot;
    const float t2v = tslot;

    v8f c = __builtin_amdgcn_wmma_f32_16x16x32_f16(
        /*neg_a=*/false, a, /*neg_b=*/false, bv,
        /*c_mod=*/(short)0, cz, /*reuse_a=*/false, /*reuse_b=*/false);

    const int rw = rn & (K_ - 1);          // wrapped prefetch row (harmless on last 2 iters)
    bslot = *(const v16h_al*)&sm_means[rw * 40 + bco_h];
    tslot = sm_t2[rw];

    float v[8];
    #pragma unroll
    for (int i = 0; i < 8; ++i) v[i] = __builtin_fmaf(c[i], LOG2E, t2v);
    float tmax = v[0];
    #pragma unroll
    for (int i = 1; i < 8; ++i) tmax = __builtin_fmaxf(tmax, v[i]);
    const float nm = __builtin_fmaxf(mrun, tmax);
    const float scale = __builtin_amdgcn_exp2f(mrun - nm);
    #pragma unroll
    for (int i = 0; i < 8; ++i)
      ssum[i] = __builtin_fmaf(ssum[i], scale, __builtin_amdgcn_exp2f(v[i] - nm));
    mrun = nm;
  };

  for (int kt = 0; kt < 32; kt += 2) {
    step(b0, t0, (kt + 2) * 16 + row);
    step(b1, t1, (kt + 3) * 16 + row);
  }

  // ---------- cross-lane merge over the 16 columns of each half-wave ----------
  #pragma unroll
  for (int off = 1; off <= 8; off <<= 1) {
    const float om = __shfl_xor(mrun, off, 32);
    const float nm = __builtin_fmaxf(mrun, om);
    const float sa = __builtin_amdgcn_exp2f(mrun - nm);
    const float sb = __builtin_amdgcn_exp2f(om   - nm);
    #pragma unroll
    for (int i = 0; i < 8; ++i) {
      const float os = __shfl_xor(ssum[i], off, 32);
      ssum[i] = ssum[i] * sa + os * sb;
    }
    mrun = nm;
  }

  // lanes 0 / 16 hold rows 0-7 / 8-15 of the tile; stage then store coalesced
  if (lane == 0 || lane == 16) {
    const int base = wid * 16 + ((lane == 16) ? 8 : 0);
    #pragma unroll
    for (int i = 0; i < 8; ++i)
      sm_stage[base + i] = mrun + __builtin_amdgcn_logf(ssum[i]);  // lse2 per row
  }
  __builtin_amdgcn_wave_barrier();   // DS ops are in-order within a wave
  if (lane < 16) {
    const float lse2 = sm_stage[wid * 16 + lane];
    const float nll  = __builtin_fmaf(0.5f, x2, LOG_NORM) + LN2 * (lse2_b - lse2);
    out[(size_t)b * S_ + (s_base + lane)] = nll;
  }
}

extern "C" void kernel_launch(void* const* d_in, const int* in_sizes, int n_in,
                              void* d_out, int out_size, void* d_ws, size_t ws_size,
                              hipStream_t stream) {
  (void)in_sizes; (void)n_in; (void)out_size; (void)d_ws; (void)ws_size;
  const float* x      = (const float*)d_in[0];
  const float* logits = (const float*)d_in[1];
  const float* means  = (const float*)d_in[2];
  float* out          = (float*)d_out;

  dim3 grid(S_ / 128, B_);   // 8 s-blocks x 256 batches
  gmm_nll_wmma_kernel<<<grid, 256, 0, stream>>>(x, logits, means, out);
}
